// MorphModel_52484500357791
// MI455X (gfx1250) — matrix operand order repaired and checked by reading, not compiled, not run on hardware
//
#include <hip/hip_runtime.h>
#include <hip/hip_bf16.h>

#define B_TOK 8192
#define D_IN  512
#define D_HID 2048
#define D_OUT 512
#define N_EXP 16
#define TOPK  2

typedef __attribute__((ext_vector_type(16))) __bf16 v16bf;
typedef __attribute__((ext_vector_type(8)))  __bf16 v8bf;
typedef __attribute__((ext_vector_type(8)))  float  v8f;

union Frag { v16bf v; v8bf h[2]; };

__device__ __forceinline__ unsigned short f2bf(float f) {
  unsigned u = __builtin_bit_cast(unsigned, f);
  u += 0x7FFFu + ((u >> 16) & 1u);          // round-to-nearest-even
  return (unsigned short)(u >> 16);
}

// ---------------- kernel 1: zero output + expert counts ----------------
__global__ void zero_kernel(float* __restrict__ out, int* __restrict__ counts, int n) {
  int i = blockIdx.x * 256 + threadIdx.x;
  if (i < n) out[i] = 0.f;
  if (blockIdx.x == 0 && threadIdx.x < N_EXP) counts[threadIdx.x] = 0;
}

// ---------------- kernel 2: gating (logits -> top2 -> softmax) ----------------
__global__ void gate_kernel(const float* __restrict__ x, const float* __restrict__ gw,
                            const float* __restrict__ gb, int* __restrict__ counts,
                            int* __restrict__ exp_idx, float* __restrict__ exp_w) {
  __shared__ float gws[D_IN * N_EXP];       // 32 KB
  for (int i = threadIdx.x; i < D_IN * N_EXP; i += 256) gws[i] = gw[i];
  __syncthreads();
  int b = blockIdx.x * 256 + threadIdx.x;
  if (b >= B_TOK) return;
  float l[N_EXP];
#pragma unroll
  for (int e = 0; e < N_EXP; ++e) l[e] = gb[e];
  const float* xr = x + (size_t)b * D_IN;
  for (int k = 0; k < D_IN; ++k) {
    float xv = xr[k];
#pragma unroll
    for (int e = 0; e < N_EXP; ++e) l[e] = fmaf(xv, gws[k * N_EXP + e], l[e]);
  }
  int i1 = 0; float v1 = l[0];
#pragma unroll
  for (int e = 1; e < N_EXP; ++e) if (l[e] > v1) { v1 = l[e]; i1 = e; }
  int i2 = 0; float v2 = -3.4e38f;
#pragma unroll
  for (int e = 0; e < N_EXP; ++e) if (e != i1 && l[e] > v2) { v2 = l[e]; i2 = e; }
  float ex = __expf(v2 - v1);
  float s = 1.f / (1.f + ex);
  exp_idx[b * 2 + 0] = i1; exp_idx[b * 2 + 1] = i2;
  exp_w[b * 2 + 0] = s;    exp_w[b * 2 + 1] = ex * s;
  atomicAdd(&counts[i1], 1);
  atomicAdd(&counts[i2], 1);
}

// ---------------- kernel 3: tiny exclusive scan over 16 counts ----------------
__global__ void scan_kernel(const int* __restrict__ counts, int* __restrict__ offsets,
                            int* __restrict__ cursors) {
  if (threadIdx.x == 0) {
    int acc = 0;
    for (int e = 0; e < N_EXP; ++e) { offsets[e] = acc; cursors[e] = acc; acc += counts[e]; }
    offsets[N_EXP] = acc;
  }
}

// ---------------- kernel 4: scatter tokens into per-expert segments ----------------
__global__ void scatter_kernel(const int* __restrict__ exp_idx, const float* __restrict__ exp_w,
                               int* __restrict__ cursors, int* __restrict__ tok_ids,
                               float* __restrict__ tok_w) {
  int t = blockIdx.x * 256 + threadIdx.x;
  if (t >= B_TOK * TOPK) return;
  int e = exp_idx[t];
  int pos = atomicAdd(&cursors[e], 1);
  tok_ids[pos] = t >> 1;
  tok_w[pos] = exp_w[t];
}

// ---- kernels 5/6: f32 weights -> bf16 in WMMA B-fragment order ----
// layout: tile (e, nt, kb) holds 512 bf16; lane L element i <-> (K=32*kb+8*(L>=16)+(i&7)+16*(i>=8), N=16*nt+(L&15))
__global__ void conv_frag_kernel(const float* __restrict__ W, unsigned short* __restrict__ Wb,
                                 int Kdim, int Ndim, int total) {
  int t = blockIdx.x * 256 + threadIdx.x;
  if (t >= total) return;
  int tile = t >> 9, r = t & 511;
  int L = r >> 4, i = r & 15;
  int ktiles = Kdim >> 5, ntiles = Ndim >> 4;
  int per_e = ktiles * ntiles;
  int e = tile / per_e, tl = tile - e * per_e;
  int nt = tl / ktiles, kb = tl - nt * ktiles;
  int Ki = (kb << 5) + ((L >> 4) << 3) + (i & 7) + ((i >> 3) << 4);
  int Ni = (nt << 4) + (L & 15);
  Wb[t] = f2bf(W[((size_t)e * Kdim + Ki) * (size_t)Ndim + Ni]);
}

// ---------------- kernel 7: fused per-expert MLP tile (16 tokens) ----------------
#define XPAD 520   // 512 + 8 (bank-shift padding, keeps 16B alignment)
#define HPAD 2056  // 2048 + 8

__global__ __launch_bounds__(256) void moe_expert_kernel(
    const float* __restrict__ x, const float* __restrict__ b1, const float* __restrict__ b2,
    const unsigned short* __restrict__ w1b, const unsigned short* __restrict__ w2b,
    const int* __restrict__ offsets, const int* __restrict__ tok_ids,
    const float* __restrict__ tok_w, float* __restrict__ out) {
  __shared__ __align__(16) unsigned short xA[16][XPAD];   // x tile, bf16
  __shared__ __align__(16) unsigned short hB[16][HPAD];   // hidden tile, bf16
  __shared__ int   toksS[16];
  __shared__ float wtsS[16];

  const int e = blockIdx.y;
  const int start = offsets[e];
  const int end = offsets[e + 1];
  const int row0 = start + blockIdx.x * 16;
  if (row0 >= end) return;                     // inactive tile: whole block exits

  const int tid = threadIdx.x;
  const int lane = tid & 31;
  const int wave = tid >> 5;                   // 8 waves
  const int lhalf = lane >> 4;
  const int lcol = lane & 15;

  if (tid < 16) {
    int idx = row0 + tid;
    int tok = (idx < end) ? tok_ids[idx] : -1;
    toksS[tid] = tok;
    wtsS[tid] = (idx < end) ? tok_w[idx] : 0.f;
  }
  __syncthreads();

  // gather x rows -> LDS bf16 (coalesced over k)
  for (int i = tid; i < 16 * D_IN; i += 256) {
    int m = i >> 9, k = i & (D_IN - 1);
    int tok = toksS[m];
    float v = (tok >= 0) ? x[(size_t)tok * D_IN + k] : 0.f;
    xA[m][k] = f2bf(v);
  }
  __syncthreads();

  // -------- phase 1: h = relu(x @ w1[e] + b1[e]); wave owns 256 hidden cols --------
  const float* b1e = b1 + (size_t)e * D_HID;
  for (int g = 0; g < 4; ++g) {
    const int nt0 = wave * 16 + g * 4;         // 4 N-tiles share each A-fragment
    v8f acc[4] = {};
    for (int kb = 0; kb < 16; ++kb) {
      Frag a;
      const unsigned short* rp = &xA[lcol][0] + (kb << 5) + (lhalf << 3);
      a.h[0] = *(const v8bf*)rp;               // K runs: {0..7|8..15}
      a.h[1] = *(const v8bf*)(rp + 16);        //          {16..23|24..31}
#pragma unroll
      for (int j = 0; j < 4; ++j) {
        Frag bb;
        const unsigned short* tb =
            w1b + ((((size_t)e * 128 + nt0 + j) << 4) + kb) * 512 + lane * 16;
        bb.h[0] = *(const v8bf*)tb;
        bb.h[1] = *(const v8bf*)(tb + 8);
        acc[j] = __builtin_amdgcn_wmma_f32_16x16x32_bf16(
            false, a.v, false, bb.v, (short)0, acc[j], false, false);
      }
    }
#pragma unroll
    for (int j = 0; j < 4; ++j) {
      int ncol = (nt0 + j) << 4;
      float bias = b1e[ncol + lcol];
#pragma unroll
      for (int r = 0; r < 8; ++r) {
        float v = acc[j][r] + bias;
        v = v > 0.f ? v : 0.f;                 // ReLU
        hB[r + (lhalf << 3)][ncol + lcol] = f2bf(v);
      }
    }
  }
  __syncthreads();

  // -------- phase 2: y = h @ w2[e] + b2[e]; scale & scatter-add --------
  const float* b2e = b2 + (size_t)e * D_OUT;
  const int nt2 = wave * 4;                    // wave owns 64 output cols
  v8f acc[4] = {};
  for (int kb = 0; kb < 64; ++kb) {
    Frag a;
    const unsigned short* rp = &hB[lcol][0] + (kb << 5) + (lhalf << 3);
    a.h[0] = *(const v8bf*)rp;
    a.h[1] = *(const v8bf*)(rp + 16);
#pragma unroll
    for (int j = 0; j < 4; ++j) {
      Frag bb;
      const unsigned short* tb =
          w2b + ((((size_t)e * 32 + nt2 + j) << 6) + kb) * 512 + lane * 16;
      bb.h[0] = *(const v8bf*)tb;
      bb.h[1] = *(const v8bf*)(tb + 8);
      acc[j] = __builtin_amdgcn_wmma_f32_16x16x32_bf16(
          false, a.v, false, bb.v, (short)0, acc[j], false, false);
    }
  }
#pragma unroll
  for (int j = 0; j < 4; ++j) {
    int ncol = (nt2 + j) << 4;
    float bias = b2e[ncol + lcol];
#pragma unroll
    for (int r = 0; r < 8; ++r) {
      int m = r + (lhalf << 3);
      int tok = toksS[m];
      if (tok >= 0) {
        float v = (acc[j][r] + bias) * wtsS[m];
        atomicAdd(&out[(size_t)tok * D_OUT + ncol + lcol], v);
      }
    }
  }
}

// ---------------- host-side launch ----------------
extern "C" void kernel_launch(void* const* d_in, const int* in_sizes, int n_in,
                              void* d_out, int out_size, void* d_ws, size_t ws_size,
                              hipStream_t stream) {
  const float* x      = (const float*)d_in[0];
  const float* gate_w = (const float*)d_in[1];
  const float* gate_b = (const float*)d_in[2];
  const float* w1     = (const float*)d_in[3];
  const float* b1     = (const float*)d_in[4];
  const float* w2     = (const float*)d_in[5];
  const float* b2     = (const float*)d_in[6];
  float* out = (float*)d_out;

  char* ws = (char*)d_ws;
  int*   counts  = (int*)(ws + 0);
  int*   offsets = (int*)(ws + 256);
  int*   cursors = (int*)(ws + 512);
  int*   exp_idx = (int*)(ws + 1024);
  float* exp_w   = (float*)(ws + 1024 + 1 * 65536);
  int*   tok_ids = (int*)(ws + 1024 + 2 * 65536);
  float* tok_w   = (float*)(ws + 1024 + 3 * 65536);
  unsigned short* w1b = (unsigned short*)(ws + 1024 + 4 * 65536);
  unsigned short* w2b = w1b + (size_t)N_EXP * D_IN * D_HID;

  const int nOut = B_TOK * D_OUT;
  zero_kernel<<<(nOut + 255) / 256, 256, 0, stream>>>(out, counts, nOut);
  gate_kernel<<<B_TOK / 256, 256, 0, stream>>>(x, gate_w, gate_b, counts, exp_idx, exp_w);
  scan_kernel<<<1, 32, 0, stream>>>(counts, offsets, cursors);
  scatter_kernel<<<(B_TOK * TOPK + 255) / 256, 256, 0, stream>>>(exp_idx, exp_w, cursors,
                                                                 tok_ids, tok_w);
  const int tw1 = N_EXP * D_IN * D_HID;
  conv_frag_kernel<<<(tw1 + 255) / 256, 256, 0, stream>>>(w1, w1b, D_IN, D_HID, tw1);
  const int tw2 = N_EXP * D_HID * D_OUT;
  conv_frag_kernel<<<(tw2 + 255) / 256, 256, 0, stream>>>(w2, w2b, D_HID, D_OUT, tw2);

  dim3 grid(B_TOK / 16, N_EXP);   // max 512 tiles/expert; inactive tiles early-exit
  moe_expert_kernel<<<grid, 256, 0, stream>>>(x, b1, b2, w1b, w2b, offsets, tok_ids,
                                              tok_w, out);
}
